// GNNEncoder_88493506167341
// MI455X (gfx1250) — matrix operand order repaired and checked by reading, not compiled, not run on hardware
//
#include <hip/hip_runtime.h>

typedef __attribute__((ext_vector_type(16))) _Float16 v16h;
typedef __attribute__((ext_vector_type(8)))  _Float16 v8h;
typedef __attribute__((ext_vector_type(8)))  float    v8f;

#define NN 50000
#define NE 600000
#define DD 128
#define HH 128

// ---------------- zero agg + deg ----------------
__global__ void __launch_bounds__(256) k_zero(float* __restrict__ agg,
                                              float* __restrict__ deg) {
  int i = blockIdx.x * 256 + threadIdx.x;
  if (i < NN * DD) agg[i] = 0.0f;
  if (i < NN) deg[i] = 0.0f;
}

// ---------------- edge scatter: agg[dst] += x[src]; deg[dst] += 1 ----------------
// one thread per (edge, 4-feature chunk): b128 gather + L2-resident f32 atomics
__global__ void __launch_bounds__(256) k_scatter(const float* __restrict__ x,
                                                 const int* __restrict__ ei,
                                                 float* __restrict__ agg,
                                                 float* __restrict__ deg) {
  int t = blockIdx.x * 256 + threadIdx.x;
  if (t >= NE * (DD / 4)) return;
  int e  = t >> 5;            // edge id (32 chunks per edge)
  int f4 = (t & 31) << 2;     // feature offset
  int src = ei[e];
  int dst = ei[NE + e];
  float4 v = *reinterpret_cast<const float4*>(x + (size_t)src * DD + f4);
  float* a = agg + (size_t)dst * DD + f4;
  atomicAdd(a + 0, v.x);
  atomicAdd(a + 1, v.y);
  atomicAdd(a + 2, v.z);
  atomicAdd(a + 3, v.w);
  if (f4 == 0) atomicAdd(deg + dst, 1.0f);
}

// ---------------- convert weights to f16 once ----------------
__global__ void __launch_bounds__(256) k_cvt(const float* __restrict__ Wl,
                                             const float* __restrict__ Wr,
                                             _Float16* __restrict__ Wl16,
                                             _Float16* __restrict__ Wr16) {
  int i = blockIdx.x * 256 + threadIdx.x;
  if (i < HH * DD) {
    Wl16[i] = (_Float16)Wl[i];
    Wr16[i] = (_Float16)Wr[i];
  }
}

// ---------------- fused dual-GEMM epilogue via WMMA ----------------
// out = (agg/max(deg,1)) @ Wl^T + b + x @ Wr^T
// block = 256 thr (8 waves); wave w -> n-tile w; blockIdx.x -> m-tile (50000 = 3125*16)
// Weights (64 KB) and the block's A tile (8 KB, pre-normalized f16) staged in LDS;
// inner loop feeds v_wmma entirely from ds_load.
__global__ void __launch_bounds__(256) k_gemm(const float* __restrict__ agg,
                                              const float* __restrict__ deg,
                                              const float* __restrict__ x,
                                              const _Float16* __restrict__ Wl16,
                                              const _Float16* __restrict__ Wr16,
                                              const float* __restrict__ bl,
                                              float* __restrict__ out) {
  __shared__ __align__(32) _Float16 sWl[HH * DD];   // 32 KB
  __shared__ __align__(32) _Float16 sWr[HH * DD];   // 32 KB
  __shared__ __align__(32) _Float16 sA1[16 * DD];   // 4 KB: agg/deg tile
  __shared__ __align__(32) _Float16 sA2[16 * DD];   // 4 KB: x tile

  // stage weights: 16384 halfs each, 8 halfs (16B) per thread per iteration
  for (int i = threadIdx.x * 8; i < HH * DD; i += 256 * 8) {
    *reinterpret_cast<float4*>(&sWl[i]) = *reinterpret_cast<const float4*>(&Wl16[i]);
    *reinterpret_cast<float4*>(&sWr[i]) = *reinterpret_cast<const float4*>(&Wr16[i]);
  }

  // stage A tile: thread t -> row r = t>>4, cols c..c+7; fold mean-normalization here
  {
    const int r = threadIdx.x >> 4;
    const int c = (threadIdx.x & 15) * 8;
    const int grow = blockIdx.x * 16 + r;
    const float id = 1.0f / fmaxf(deg[grow], 1.0f);
    const float* __restrict__ ar = agg + (size_t)grow * DD + c;
    const float* __restrict__ xr = x   + (size_t)grow * DD + c;
    v8h h1, h2;
#pragma unroll
    for (int i = 0; i < 8; ++i) {
      h1[i] = (_Float16)(ar[i] * id);
      h2[i] = (_Float16)(xr[i]);
    }
    *reinterpret_cast<v8h*>(&sA1[r * DD + c]) = h1;
    *reinterpret_cast<v8h*>(&sA2[r * DD + c]) = h2;
  }
  __syncthreads();

  const int lane = threadIdx.x & 31;
  const int wv   = threadIdx.x >> 5;        // n-tile index 0..7
  const int hiK  = lane >> 4;               // lane group 0/1
  const int l16  = lane & 15;
  const int nCol = wv * 16 + l16;           // output column (N)

  const _Float16* __restrict__ a1Row = &sA1[l16 * DD];
  const _Float16* __restrict__ a2Row = &sA2[l16 * DD];
  const _Float16* __restrict__ wlRow = &sWl[nCol * DD];
  const _Float16* __restrict__ wrRow = &sWr[nCol * DD];

  v8f acc = {};
#pragma unroll
  for (int kt = 0; kt < DD; kt += 32) {
    const int kb = kt + hiK * 16;  // A layout: 16 contiguous halves K=kb..kb+15 (matches B)
    const v8h lo1 = *reinterpret_cast<const v8h*>(a1Row + kb);
    const v8h hi1 = *reinterpret_cast<const v8h*>(a1Row + kb + 8);
    const v8h lo2 = *reinterpret_cast<const v8h*>(a2Row + kb);
    const v8h hi2 = *reinterpret_cast<const v8h*>(a2Row + kb + 8);
    const v16h a1 = __builtin_shufflevector(lo1, hi1, 0,1,2,3,4,5,6,7,8,9,10,11,12,13,14,15);
    const v16h a2 = __builtin_shufflevector(lo2, hi2, 0,1,2,3,4,5,6,7,8,9,10,11,12,13,14,15);
    // B layout used here (32x16, 16-bit, contiguous k, same as A): lanes 0-15 hold K=kt..kt+15, lanes 16-31 K=kt+16..kt+31
    const v16h b1 = *reinterpret_cast<const v16h*>(wlRow + kt + hiK * 16);
    const v16h b2 = *reinterpret_cast<const v16h*>(wrRow + kt + hiK * 16);
    acc = __builtin_amdgcn_wmma_f32_16x16x32_f16(false, a1, false, b1, (short)0, acc, false, false);
    acc = __builtin_amdgcn_wmma_f32_16x16x32_f16(false, a2, false, b2, (short)0, acc, false, false);
  }

  const float bias = bl[nCol];
  // C/D layout: VGPR r -> M = hiK*8 + r, N = l16
#pragma unroll
  for (int r = 0; r < 8; ++r) {
    out[(size_t)(blockIdx.x * 16 + hiK * 8 + r) * HH + nCol] = acc[r] + bias;
  }
}

extern "C" void kernel_launch(void* const* d_in, const int* in_sizes, int n_in,
                              void* d_out, int out_size, void* d_ws, size_t ws_size,
                              hipStream_t stream) {
  const float* x  = (const float*)d_in[0];
  const int*   ei = (const int*)d_in[1];   // edge_index [2, E]: row0=src, row1=dst
  const float* Wl = (const float*)d_in[2];
  const float* bl = (const float*)d_in[3];
  const float* Wr = (const float*)d_in[4];
  float* out = (float*)d_out;

  char* ws = (char*)d_ws;
  float*    agg  = (float*)ws;                                         // NN*DD f32 = 25.6 MB
  float*    deg  = (float*)(ws + (size_t)NN * DD * 4);                 // NN f32
  _Float16* Wl16 = (_Float16*)(ws + (size_t)NN * DD * 4 + NN * 4);     // 32 KB
  _Float16* Wr16 = Wl16 + HH * DD;                                     // 32 KB

  k_zero   <<<(NN * DD + 255) / 256, 256, 0, stream>>>(agg, deg);
  k_scatter<<<(NE * (DD / 4) + 255) / 256, 256, 0, stream>>>(x, ei, agg, deg);
  k_cvt    <<<(HH * DD + 255) / 256, 256, 0, stream>>>(Wl, Wr, Wl16, Wr16);
  k_gemm   <<<NN / 16, 256, 0, stream>>>(agg, deg, x, Wl16, Wr16, bl, out);
}
